// ResidualMamba3_44848048505419
// MI455X (gfx1250) — matrix operand order repaired and checked by reading, not compiled, not run on hardware
//
#include <hip/hip_runtime.h>
#include <hip/hip_bf16.h>
#include <stdint.h>

// ---------------------------------------------------------------------------
// ResidualMamba3 for MI455X (gfx1250, wave32).
// GEMMs: v_wmma_f32_16x16x32_bf16, 64x32 per-wave tiles (8 WMMA / K-step,
// 1.5 b128 loads per WMMA). B tiles are block-shared -> staged via
// global_load_async_to_lds_b128 (ASYNCcnt) with double buffering when the
// builtin exists; direct-global fallback otherwise. Operands are L2-resident
// (192MB L2), so the win is WGP$/L2 bandwidth, not HBM.
// Scan: sequential VALU, register state, double-buffered LDS staging.
// ---------------------------------------------------------------------------

typedef __attribute__((ext_vector_type(16))) __bf16 v16bf;
typedef __attribute__((ext_vector_type(8)))  float  v8f;

union BFrag { v16bf v; uint4 u[2]; };

static __device__ __forceinline__ unsigned short f32_to_bf16_rne(float f) {
    unsigned int u = __float_as_uint(f);
    u += 0x7FFFu + ((u >> 16) & 1u);   // round to nearest even
    return (unsigned short)(u >> 16);
}
static __device__ __forceinline__ float silu(float x) {
    return x / (1.f + __expf(-x));
}
static __device__ __forceinline__ v8f wmma_bf16(const BFrag& a, const BFrag& b, v8f c) {
    return __builtin_amdgcn_wmma_f32_16x16x32_bf16(false, a.v, false, b.v,
                                                   (short)0, c, false, false);
}

// --- async global->LDS staging (gfx1250, ASYNCcnt) with compile-time probe --
#if __has_builtin(__builtin_amdgcn_global_load_async_to_lds_b128) && \
    __has_builtin(__builtin_amdgcn_s_wait_asynccnt)
#define HAVE_ASYNC_LDS 1
typedef int v4i_vs __attribute__((vector_size(4 * sizeof(int))));
typedef __attribute__((address_space(1))) v4i_vs* g1_v4i_p;   // global (__device__)
typedef __attribute__((address_space(3))) v4i_vs* l3_v4i_p;   // LDS (__shared__)
static __device__ __forceinline__ void async_copy_b128(const void* g, void* l) {
    // flat->LDS: low 32 bits of a flat shared address are the LDS byte offset
    __builtin_amdgcn_global_load_async_to_lds_b128(
        (g1_v4i_p)(uintptr_t)g, (l3_v4i_p)(unsigned int)(uintptr_t)l, 0, 0);
}
#else
#define HAVE_ASYNC_LDS 0
#endif

// ----------------------- constants -----------------------------------------
#define D_MODEL   1024
#define D_STATE   128
#define HEADDIM   64
#define D_INNER   2048
#define NHEADS    32
#define CONV_DIM  2304
#define D_IN_PROJ 4384
#define SEQLEN    2048
#define NROWS     8192          // BATCH*SEQLEN

// ----------------------- bf16 packing kernels ------------------------------
__global__ __launch_bounds__(256) void pack_winT(const float* __restrict__ W,
                                                 unsigned short* __restrict__ o) {
    int idx = blockIdx.x * 256 + threadIdx.x;          // 4384*1024
    int n = idx >> 10, k = idx & 1023;
    o[idx] = f32_to_bf16_rne(W[(size_t)k * D_IN_PROJ + n]);
}
__global__ __launch_bounds__(256) void pack_woutT(const float* __restrict__ W,
                                                  unsigned short* __restrict__ o) {
    int idx = blockIdx.x * 256 + threadIdx.x;          // 1024*2048
    int n = idx >> 11, k = idx & 2047;
    o[idx] = f32_to_bf16_rne(W[(size_t)k * D_MODEL + n]);
}
__global__ __launch_bounds__(256) void pack_x(const float* __restrict__ x,
                                              unsigned short* __restrict__ o) {
    int idx = blockIdx.x * 256 + threadIdx.x;          // 8192*1024
    o[idx] = f32_to_bf16_rne(x[idx]);
}

// ----------------------- bf16 GEMM, 64x32 per-wave tile --------------------
// A: bf16 row-major (NROWS x K). BT: bf16 N-major (N x K).
// Block = 8 waves; wave w owns rows [ (by*8+w)*64, +64 ), block owns 32 cols.
// B tile (32 cols x 32 K, 2KB) is identical for all 8 waves -> LDS staged.
template<int K, int NOUT, bool RESID>
__global__ __launch_bounds__(256) void gemm_bf16(const unsigned short* __restrict__ A,
                                                 const unsigned short* __restrict__ BT,
                                                 const float* __restrict__ resid,
                                                 float* __restrict__ C) {
    const int tid  = threadIdx.x;
    const int wave = tid >> 5, lane = tid & 31;
    const int lm   = lane & 15;
    const int sel  = lane >> 4;
    const int n16  = blockIdx.x * 32;
    const int m64  = (blockIdx.y * 8 + wave) * 64;

    const unsigned short* ap[4];
    #pragma unroll
    for (int i = 0; i < 4; ++i) ap[i] = A + (size_t)(m64 + i * 16 + lm) * K;

    v8f c[4][2];
    #pragma unroll
    for (int i = 0; i < 4; ++i) { c[i][0] = v8f{}; c[i][1] = v8f{}; }

#if HAVE_ASYNC_LDS
    // double-buffered B tile: [buf][n_local 0..31][k_local 0..31] bf16
    __shared__ unsigned short sB[2][32 * 32];
    const int nl = tid >> 2, ch = tid & 3;           // threads 0..127 stage
    const unsigned short* bstage = BT + (size_t)(n16 + nl) * K + ch * 8;

    if (tid < 128) async_copy_b128(bstage, &sB[0][nl * 32 + ch * 8]);

    int buf = 0;
    for (int k0 = 0; k0 < K; k0 += 32) {
        __builtin_amdgcn_s_wait_asynccnt(0);
        __syncthreads();                              // B[buf] visible to all
        if (k0 + 32 < K && tid < 128)
            async_copy_b128(bstage + k0 + 32, &sB[buf ^ 1][nl * 32 + ch * 8]);

        BFrag b0, b1;
        b0.u[0] = *(const uint4*)&sB[buf][lm * 32 + sel * 16];
        b0.u[1] = *(const uint4*)&sB[buf][lm * 32 + sel * 16 + 8];
        b1.u[0] = *(const uint4*)&sB[buf][(lm + 16) * 32 + sel * 16];
        b1.u[1] = *(const uint4*)&sB[buf][(lm + 16) * 32 + sel * 16 + 8];

        #pragma unroll
        for (int i = 0; i < 4; ++i) {
            __builtin_prefetch(ap[i] + k0 + 128, 0, 1);
            BFrag a;
            a.u[0] = *(const uint4*)(ap[i] + k0 + sel * 16);
            a.u[1] = *(const uint4*)(ap[i] + k0 + sel * 16 + 8);
            c[i][0] = wmma_bf16(a, b0, c[i][0]);
            c[i][1] = wmma_bf16(a, b1, c[i][1]);
        }
        buf ^= 1;
    }
#else
    const unsigned short* b0p = BT + (size_t)(n16 + lm) * K;
    const unsigned short* b1p = BT + (size_t)(n16 + 16 + lm) * K;
    for (int k0 = 0; k0 < K; k0 += 32) {
        __builtin_prefetch(b0p + k0 + 128, 0, 1);
        BFrag b0, b1;
        b0.u[0] = *(const uint4*)(b0p + k0 + sel * 16);
        b0.u[1] = *(const uint4*)(b0p + k0 + sel * 16 + 8);
        b1.u[0] = *(const uint4*)(b1p + k0 + sel * 16);
        b1.u[1] = *(const uint4*)(b1p + k0 + sel * 16 + 8);
        #pragma unroll
        for (int i = 0; i < 4; ++i) {
            __builtin_prefetch(ap[i] + k0 + 128, 0, 1);
            BFrag a;
            a.u[0] = *(const uint4*)(ap[i] + k0 + sel * 16);
            a.u[1] = *(const uint4*)(ap[i] + k0 + sel * 16 + 8);
            c[i][0] = wmma_bf16(a, b0, c[i][0]);
            c[i][1] = wmma_bf16(a, b1, c[i][1]);
        }
    }
#endif

    const int col = n16 + lm;
    #pragma unroll
    for (int i = 0; i < 4; ++i) {
        const int rbase = m64 + i * 16 + sel * 8;
        #pragma unroll
        for (int r = 0; r < 8; ++r) {
            size_t i0 = (size_t)(rbase + r) * NOUT + col;
            if (RESID) {
                C[i0]      = resid[i0]      + c[i][0][r];
                C[i0 + 16] = resid[i0 + 16] + c[i][1][r];
            } else {
                C[i0]      = c[i][0][r];
                C[i0 + 16] = c[i][1][r];
            }
        }
    }
}

// ----------------------- depthwise conv(4) + SiLU --------------------------
__global__ __launch_bounds__(256) void conv_kernel(const float* __restrict__ zx,
                                                   const float* __restrict__ cw,
                                                   const float* __restrict__ cb,
                                                   float* __restrict__ xc) {
    int idx = blockIdx.x * 256 + threadIdx.x;          // 8192*2304
    int row = idx / CONV_DIM;
    int c   = idx - row * CONV_DIM;
    int l   = row & (SEQLEN - 1);
    float acc = cb[c];
    #pragma unroll
    for (int k = 0; k < 4; ++k) {
        int dl = l - 3 + k;
        if (dl >= 0)
            acc += cw[k * CONV_DIM + c] * zx[(size_t)(row - 3 + k) * D_IN_PROJ + D_INNER + c];
    }
    xc[idx] = silu(acc);
}

// ----------------------- dt softplus + dA ----------------------------------
__global__ __launch_bounds__(256) void dt_kernel(const float* __restrict__ zx,
                                                 const float* __restrict__ dt_bias,
                                                 const float* __restrict__ A_log,
                                                 float* __restrict__ dtb,
                                                 float* __restrict__ dAb) {
    int idx = blockIdx.x * 256 + threadIdx.x;          // 8192*32
    int row = idx >> 5, h = idx & 31;
    float v  = zx[(size_t)row * D_IN_PROJ + (D_INNER + CONV_DIM) + h] + dt_bias[h];
    float sp = (v > 20.f) ? v : log1pf(__expf(v));
    dtb[idx] = sp;
    dAb[idx] = __expf(-__expf(A_log[h]) * sp);
}

// ----------------------- sequential SSM scan -------------------------------
// One block per (batch, head). State h[64][128] in registers:
// thread owns p = wave*8 + (lane&7), n-chunk = (lane>>3)*32 .. +32.
// Double-buffered LDS staging: t+1 global loads issued before the step-t
// compute so their latency hides behind the 96-FMA body. One barrier/step.
__global__ __launch_bounds__(256) void scan_kernel(const float* __restrict__ xconv,
                                                   const float* __restrict__ dtb,
                                                   const float* __restrict__ dAb,
                                                   const float* __restrict__ Dp,
                                                   float* __restrict__ y) {
    const int bh = blockIdx.x;                 // 0..127
    const int b  = bh >> 5, h = bh & 31;
    const int tid = threadIdx.x;
    const int w = tid >> 5, l = tid & 31;
    const int p  = w * 8 + (l & 7);
    const int n0 = (l >> 3) * 32;
    const float Dh = Dp[h];

    float hs[32];
    #pragma unroll
    for (int j = 0; j < 32; ++j) hs[j] = 0.f;

    __shared__ float sXh[2][HEADDIM];
    __shared__ float sBC[2][2 * D_STATE];
    __shared__ float sdt[2], sdA[2];

    const float* xrow = xconv + (size_t)b * SEQLEN * CONV_DIM;

    // preload step 0
    float rBC = xrow[D_INNER + tid];
    float rXh = (tid < HEADDIM) ? xrow[h * HEADDIM + tid] : 0.f;
    float rdt = 0.f, rdA = 0.f;
    if (tid == 0) {
        rdt = dtb[(size_t)(b * SEQLEN) * NHEADS + h];
        rdA = dAb[(size_t)(b * SEQLEN) * NHEADS + h];
    }

    int buf = 0;
    for (int t = 0; t < SEQLEN; ++t) {
        sBC[buf][tid] = rBC;
        if (tid < HEADDIM) sXh[buf][tid] = rXh;
        if (tid == 0) { sdt[buf] = rdt; sdA[buf] = rdA; }
        __syncthreads();

        if (t + 1 < SEQLEN) {                       // prefetch next step
            const float* xr1 = xrow + (size_t)(t + 1) * CONV_DIM;
            rBC = xr1[D_INNER + tid];
            if (tid < HEADDIM) rXh = xr1[h * HEADDIM + tid];
            if (tid == 0) {
                int row = b * SEQLEN + t + 1;
                rdt = dtb[(size_t)row * NHEADS + h];
                rdA = dAb[(size_t)row * NHEADS + h];
            }
        }

        const float xp   = sXh[buf][p];
        const float coef = sdt[buf] * xp;
        const float dA   = sdA[buf];
        float acc = 0.f;
        #pragma unroll
        for (int j = 0; j < 32; ++j) {
            float hv = fmaf(dA, hs[j], coef * sBC[buf][n0 + j]);
            hs[j] = hv;
            acc = fmaf(hv, sBC[buf][D_STATE + n0 + j], acc);
        }
        acc += __shfl_xor(acc, 8);
        acc += __shfl_xor(acc, 16);
        if ((l >> 3) == 0) {
            int row = b * SEQLEN + t;
            y[(size_t)row * D_INNER + h * HEADDIM + p] = acc + Dh * xp;
        }
        buf ^= 1;
    }
}

// ----------------------- gate * silu(z), RMSNorm, -> bf16 ------------------
__global__ __launch_bounds__(256) void gate_rms(const float* __restrict__ y,
                                                const float* __restrict__ zx,
                                                const float* __restrict__ nw,
                                                unsigned short* __restrict__ ybf) {
    const int row = blockIdx.x, tid = threadIdx.x;
    float v[8], ss = 0.f;
    #pragma unroll
    for (int i = 0; i < 8; ++i) {
        int c = tid + i * 256;
        float z  = zx[(size_t)row * D_IN_PROJ + c];
        float yy = y[(size_t)row * D_INNER + c] * silu(z);
        v[i] = yy;
        ss = fmaf(yy, yy, ss);
    }
    __shared__ float red[256];
    red[tid] = ss;
    __syncthreads();
    for (int s = 128; s > 0; s >>= 1) {
        if (tid < s) red[tid] += red[tid + s];
        __syncthreads();
    }
    const float scale = rsqrtf(red[0] * (1.f / D_INNER) + 1e-5f);
    #pragma unroll
    for (int i = 0; i < 8; ++i) {
        int c = tid + i * 256;
        ybf[(size_t)row * D_INNER + c] = f32_to_bf16_rne(v[i] * scale * nw[c]);
    }
}

// ----------------------- host orchestration --------------------------------
extern "C" void kernel_launch(void* const* d_in, const int* in_sizes, int n_in,
                              void* d_out, int out_size, void* d_ws, size_t ws_size,
                              hipStream_t stream) {
    const float* x       = (const float*)d_in[0];
    const float* W_in    = (const float*)d_in[1];
    const float* conv_w  = (const float*)d_in[2];
    const float* conv_b  = (const float*)d_in[3];
    const float* dt_bias = (const float*)d_in[4];
    const float* A_log   = (const float*)d_in[5];
    const float* Dvec    = (const float*)d_in[6];
    const float* norm_w  = (const float*)d_in[7];
    const float* W_out   = (const float*)d_in[8];
    float* out = (float*)d_out;

    char* ws = (char*)d_ws;
    // byte offsets (all 256-aligned)
    unsigned short* winT  = (unsigned short*)(ws + 0);           // 4384x1024 bf16 (8.98 MB)
    unsigned short* woutT = (unsigned short*)(ws + 8978432);     // 1024x2048 bf16 (4.19 MB)
    unsigned short* xbf   = (unsigned short*)(ws + 13172736);    // 8192x1024 bf16 (16.8 MB)
    float* zx    = (float*)(ws + 29949952);                      // 8192x4384 f32 (143.7 MB)
    float* xconv = (float*)(ws + 173604864);                     // 8192x2304 f32 (75.5 MB)
    float* dtb   = (float*)(ws + 249102336);                     // 8192x32 f32
    float* dAb   = (float*)(ws + 250150912);                     // 8192x32 f32
    float* ybuf  = (float*)(ws + 251199488);                     // 8192x2048 f32 (67.1 MB)
    unsigned short* ybf = (unsigned short*)(ws + 173604864);     // aliases xconv (dead post-scan)

    // 1. pack bf16 operands
    pack_winT <<<dim3(17536), dim3(256), 0, stream>>>(W_in,  winT);
    pack_woutT<<<dim3(8192),  dim3(256), 0, stream>>>(W_out, woutT);
    pack_x    <<<dim3(32768), dim3(256), 0, stream>>>(x,     xbf);
    // 2. in_proj GEMM: N=4384 -> 137 col-tiles of 32; M=8192 -> 16 blocks x 8 waves x 64 rows
    gemm_bf16<1024, D_IN_PROJ, false><<<dim3(137, 16), dim3(256), 0, stream>>>(xbf, winT, nullptr, zx);
    // 3. conv + silu
    conv_kernel<<<dim3(73728), dim3(256), 0, stream>>>(zx, conv_w, conv_b, xconv);
    // 4. dt softplus / dA
    dt_kernel<<<dim3(1024), dim3(256), 0, stream>>>(zx, dt_bias, A_log, dtb, dAb);
    // 5. sequential scan, one block per (batch, head)
    scan_kernel<<<dim3(128), dim3(256), 0, stream>>>(xconv, dtb, dAb, Dvec, ybuf);
    // 6. gate + RMSNorm -> bf16
    gate_rms<<<dim3(8192), dim3(256), 0, stream>>>(ybuf, zx, norm_w, ybf);
    // 7. out_proj GEMM + residual: N=1024 -> 32 col-tiles of 32
    gemm_bf16<D_INNER, D_MODEL, true><<<dim3(32, 16), dim3(256), 0, stream>>>(ybf, woutT, x, out);
}
